// MistralAttention_18846316495166
// MI455X (gfx1250) — compile-verified, matrix operand order checked by reading, not run on hardware
//
#include <hip/hip_runtime.h>
#include <hip/hip_bf16.h>

#define S_LEN   2048
#define HIDDEN  4096
#define NH      32
#define NKV     8
#define HD      128
#define QK_SCALE 0.08838834764831845f  // 1/sqrt(128)

typedef _Float16 v16h __attribute__((ext_vector_type(16)));
typedef _Float16 v8h  __attribute__((ext_vector_type(8)));
typedef float    v8f  __attribute__((ext_vector_type(8)));

__device__ __forceinline__ v8f wmma16x16x32(v16h a, v16h b, v8f c) {
  // D = A(16x32 f16) x B(32x16 f16) + C(16x16 f32)
  return __builtin_amdgcn_wmma_f32_16x16x32_f16(false, a, false, b, (short)0, c,
                                                false, false);
}

// Load one 16-half WMMA fragment for a lane whose matrix row (A: m, B: n)
// starts at rowp (element k=0 of the current 32-wide K slice).
// ISA layout: halves 0..7 <-> k = h*8 .. h*8+7 ; halves 8..15 <-> k+16.
// => two 16-byte loads per lane (global_load_b128 / ds_load_b128).
__device__ __forceinline__ v16h load_frag(const _Float16* rowp, int h) {
  const v8h* p = (const v8h*)(rowp + (h << 3));
  v8h lo = p[0];
  v8h hi = p[2];  // +16 halves
  v16h r;
#pragma unroll
  for (int i = 0; i < 8; ++i) { r[i] = lo[i]; r[i + 8] = hi[i]; }
  return r;
}

// CDNA5 async copy: 16B global -> LDS per lane, tracked by ASYNCcnt.
// lds_off = low 32 bits of the generic LDS address (ISA 10.2: addr[31:0]).
__device__ __forceinline__ void async_b128(uint32_t lds_off, const void* gaddr) {
  asm volatile("global_load_async_to_lds_b128 %0, %1, off"
               :: "v"(lds_off), "v"(gaddr)
               : "memory");
}
__device__ __forceinline__ void wait_asynccnt0() {
  asm volatile("s_wait_asynccnt 0x0" ::: "memory");
}

// ---------------- conversion kernels ----------------

__global__ void cvt_f32_f16(const float* __restrict__ src,
                            _Float16* __restrict__ dst, int n) {
  int i = blockIdx.x * blockDim.x + threadIdx.x;
  if (i < n) dst[i] = (_Float16)src[i];
}

// src is [K][N] row-major fp32; dst is [N][K] row-major f16 (transposed)
__global__ void cvt_transpose_f32_f16(const float* __restrict__ src,
                                      _Float16* __restrict__ dst,
                                      int K, int N) {
  int i = blockIdx.x * blockDim.x + threadIdx.x;
  int total = K * N;
  if (i >= total) return;
  int k = i / N, n = i % N;         // coalesced read
  dst[(size_t)n * K + k] = (_Float16)src[i];
}

// Vh[s][kv*128+d] -> Vt[kv][d][s]
__global__ void transpose_v(const _Float16* __restrict__ Vh,
                            _Float16* __restrict__ Vt) {
  int i = blockIdx.x * blockDim.x + threadIdx.x;
  if (i >= S_LEN * NKV * HD) return;
  int s = i / (NKV * HD);
  int c = i % (NKV * HD);
  int kv = c / HD, d = c % HD;
  Vt[((size_t)kv * HD + d) * S_LEN + s] = Vh[i];
}

// in-place RoPE on f16 [S][nheads*128]; position = row index
__global__ void rope_kernel(_Float16* __restrict__ X, int nheads) {
  int i = blockIdx.x * blockDim.x + threadIdx.x;
  int total = S_LEN * nheads * 64;
  if (i >= total) return;
  int d = i % 64;
  int hh = (i / 64) % nheads;
  int s = i / (64 * nheads);
  float invf = __powf(10000.0f, -(float)d / 64.0f);
  float ang = (float)s * invf;
  float c = __cosf(ang), sn = __sinf(ang);
  _Float16* p = X + (size_t)s * (nheads * HD) + hh * HD;
  float x0 = (float)p[d], x1 = (float)p[d + 64];
  p[d]      = (_Float16)(x0 * c - x1 * sn);
  p[d + 64] = (_Float16)(x1 * c + x0 * sn);
}

// ---------------- WMMA GEMM: C[M,N] = A[M,K] * BT[N,K]^T ----------------
// 256 threads = 8 waves; wave tile 32x32 (2x2 WMMA tiles); WG tile 64x128.
template <int STOREF32>
__global__ __launch_bounds__(256) void gemm_f16_wmma(
    const _Float16* __restrict__ A, const _Float16* __restrict__ BT,
    void* __restrict__ Cout, int M, int N, int K) {
  int wave = threadIdx.x >> 5;
  int lane = threadIdx.x & 31;
  int idx = lane & 15, h = lane >> 4;
  int wrow = wave >> 2, wcol = wave & 3;
  int R  = blockIdx.y * 64  + wrow * 32;
  int Cb = blockIdx.x * 128 + wcol * 32;

  v8f acc00 = {0,0,0,0,0,0,0,0}, acc01 = acc00, acc10 = acc00, acc11 = acc00;

  const _Float16* a0p = A  + (size_t)(R  + idx)      * K;
  const _Float16* a1p = A  + (size_t)(R  + 16 + idx) * K;
  const _Float16* b0p = BT + (size_t)(Cb + idx)      * K;
  const _Float16* b1p = BT + (size_t)(Cb + 16 + idx) * K;

  for (int k = 0; k < K; k += 32) {
    __builtin_prefetch(a0p + k + 256, 0, 0);   // global_prefetch_b8
    __builtin_prefetch(b0p + k + 256, 0, 0);
    v16h a0 = load_frag(a0p + k, h);
    v16h a1 = load_frag(a1p + k, h);
    v16h b0 = load_frag(b0p + k, h);
    v16h b1 = load_frag(b1p + k, h);
    acc00 = wmma16x16x32(a0, b0, acc00);
    acc01 = wmma16x16x32(a0, b1, acc01);
    acc10 = wmma16x16x32(a1, b0, acc10);
    acc11 = wmma16x16x32(a1, b1, acc11);
  }

#pragma unroll
  for (int r = 0; r < 8; ++r) {
    int row0 = R + r + 8 * h, row1 = row0 + 16;
    int c0 = Cb + idx, c1 = c0 + 16;
    if (STOREF32) {
      float* C = (float*)Cout;
      C[(size_t)row0 * N + c0] = acc00[r];
      C[(size_t)row0 * N + c1] = acc01[r];
      C[(size_t)row1 * N + c0] = acc10[r];
      C[(size_t)row1 * N + c1] = acc11[r];
    } else {
      _Float16* C = (_Float16*)Cout;
      C[(size_t)row0 * N + c0] = (_Float16)acc00[r];
      C[(size_t)row0 * N + c1] = (_Float16)acc01[r];
      C[(size_t)row1 * N + c0] = (_Float16)acc10[r];
      C[(size_t)row1 * N + c1] = (_Float16)acc11[r];
    }
  }
}

// ---------------- flash attention (causal, GQA) ----------------
// grid (NH, S/64), block 128 (4 waves). Each wave: one 16-query tile.
// K/V blocks are shared by all 4 waves (same head) -> staged once per WG
// into LDS with async global->LDS copies, double-buffered.
__global__ __launch_bounds__(128) void flash_attn(
    const _Float16* __restrict__ Qh,   // [S][NH*HD], rope'd
    const _Float16* __restrict__ Kh,   // [S][NKV*HD], rope'd
    const _Float16* __restrict__ Vt,   // [NKV][HD][S]
    _Float16* __restrict__ Oh) {       // [S][NH*HD]
  __shared__ _Float16 ldsK[2][32 * HD];   // [buf][key][d]  8KB each
  __shared__ _Float16 ldsV[2][HD * 32];   // [buf][d][key]  8KB each
  __shared__ _Float16 lds_p[4][16 * 32];  // per-wave P transpose staging

  int head = blockIdx.x;
  int tid = threadIdx.x;
  int wave = tid >> 5;
  int lane = tid & 31;
  int idx = lane & 15, h = lane >> 4;
  int qbase = blockIdx.y * 64 + wave * 16;
  int kvh = head >> 2;  // N_REP = 4

  const _Float16* Kbase = Kh + (size_t)kvh * HD;            // + key*(NKV*HD)
  const _Float16* Vbase = Vt + (size_t)kvh * HD * S_LEN;    // + d*S_LEN

  // Q fragments: 4 chunks of K=32 covering head_dim=128
  const _Float16* qrow = Qh + (size_t)(qbase + idx) * (NH * HD) + head * HD;
  v16h aq[4];
#pragma unroll
  for (int t = 0; t < 4; ++t) aq[t] = load_frag(qrow + t * 32, h);

  v8f o[8];
#pragma unroll
  for (int t = 0; t < 8; ++t) o[t] = (v8f){0,0,0,0,0,0,0,0};
  float mrow[8], lrow[8];
#pragma unroll
  for (int r = 0; r < 8; ++r) { mrow[r] = -3.0e38f; lrow[r] = 0.0f; }

  // All 4 waves iterate the same range (cooperative staging + barriers);
  // tiles past a wave's causal limit are killed by the -1e9 mask.
  int kbmax = 2 * blockIdx.y + 1;

  // issue async copy of key-block kb into buffer buf (8KB K + 8KB V)
  auto issue_block = [&](int kb, int buf) {
    int k0 = kb * 32;
    // K tile: 32 keys x 128 halves, 16B chunks, 4 per lane
#pragma unroll
    for (int c = 0; c < 4; ++c) {
      int chunk = tid + c * 128;           // 0..511
      int key = chunk >> 4;                // 16 chunks per 256B row
      int off = (chunk & 15) << 3;         // halves
      async_b128((uint32_t)(uintptr_t)&ldsK[buf][key * HD + off],
                 Kbase + (size_t)(k0 + key) * (NKV * HD) + off);
    }
    // V tile: 128 d-rows x 32 keys, 4 chunks per d-row
#pragma unroll
    for (int c = 0; c < 4; ++c) {
      int chunk = tid + c * 128;
      int d = chunk >> 2;
      int off = (chunk & 3) << 3;
      async_b128((uint32_t)(uintptr_t)&ldsV[buf][d * 32 + off],
                 Vbase + (size_t)d * S_LEN + k0 + off);
    }
  };

  issue_block(0, 0);

  for (int kb = 0; kb <= kbmax; ++kb) {
    int cur = kb & 1;
    wait_asynccnt0();        // buf[cur] landed in LDS
    __syncthreads();         // visible to all waves
    if (kb < kbmax) issue_block(kb + 1, cur ^ 1);

    // scores: two 16-key subtiles, K fragments from LDS (ds_load_b128)
    v8f sc[2];
#pragma unroll
    for (int j = 0; j < 2; ++j) {
      int k0 = kb * 32 + j * 16;
      v8f s = {0,0,0,0,0,0,0,0};
      const _Float16* krow = &ldsK[cur][(j * 16 + idx) * HD];
#pragma unroll
      for (int t = 0; t < 4; ++t) {
        v16h bk = load_frag(krow + t * 32, h);
        s = wmma16x16x32(aq[t], bk, s);
      }
#pragma unroll
      for (int r = 0; r < 8; ++r) {
        int qg = qbase + r + 8 * h;
        int kg = k0 + idx;
        float v = s[r] * QK_SCALE;
        sc[j][r] = (kg > qg) ? -1.0e9f : v;
      }
    }

    // online softmax: each row's 16 cols live across the 16-lane half-wave
    float al[8];
#pragma unroll
    for (int r = 0; r < 8; ++r) {
      float mx = fmaxf(sc[0][r], sc[1][r]);
#pragma unroll
      for (int d = 1; d < 16; d <<= 1) mx = fmaxf(mx, __shfl_xor(mx, d, 32));
      float mnew = fmaxf(mrow[r], mx);
      float alpha = __expf(mrow[r] - mnew);
      float p0 = __expf(sc[0][r] - mnew);
      float p1 = __expf(sc[1][r] - mnew);
      float sum = p0 + p1;
#pragma unroll
      for (int d = 1; d < 16; d <<= 1) sum += __shfl_xor(sum, d, 32);
      lrow[r] = lrow[r] * alpha + sum;
      mrow[r] = mnew;
      al[r] = alpha;
      int row = r + 8 * h;
      lds_p[wave][row * 32 + idx]      = (_Float16)p0;  // C-layout -> LDS
      lds_p[wave][row * 32 + 16 + idx] = (_Float16)p1;
    }
#pragma unroll
    for (int t = 0; t < 8; ++t)
#pragma unroll
      for (int r = 0; r < 8; ++r) o[t][r] *= al[r];

    // P back out of LDS in A-layout (same-wave region, no barrier needed)
    v16h ap = load_frag(&lds_p[wave][idx * 32], h);

    // PV: V fragments from LDS [d][key], contiguous along keys
#pragma unroll
    for (int t = 0; t < 8; ++t) {
      v16h bv = load_frag(&ldsV[cur][(t * 16 + idx) * 32], h);
      o[t] = wmma16x16x32(ap, bv, o[t]);
    }

    __syncthreads();  // everyone done with buf[cur] before it is reissued
  }

#pragma unroll
  for (int t = 0; t < 8; ++t)
#pragma unroll
    for (int r = 0; r < 8; ++r) {
      int q = qbase + r + 8 * h;
      int col = head * HD + t * 16 + idx;
      Oh[(size_t)q * (NH * HD) + col] = (_Float16)(o[t][r] / lrow[r]);
    }
}

// ---------------- host orchestration ----------------

extern "C" void kernel_launch(void* const* d_in, const int* in_sizes, int n_in,
                              void* d_out, int out_size, void* d_ws,
                              size_t ws_size, hipStream_t stream) {
  const float* X  = (const float*)d_in[0];   // [2048][4096]
  const float* Wq = (const float*)d_in[1];   // [4096][4096]
  const float* Wk = (const float*)d_in[2];   // [4096][1024]
  const float* Wv = (const float*)d_in[3];   // [4096][1024]
  const float* Wo = (const float*)d_in[4];   // [4096][4096]
  // d_in[5] mask, d_in[6] position_ids: implied by causal + arange

  _Float16* ws = (_Float16*)d_ws;
  size_t off = 0;
  _Float16* Xh  = ws + off; off += (size_t)S_LEN * HIDDEN;        // 8M
  _Float16* WqT = ws + off; off += (size_t)HIDDEN * (NH * HD);    // 16M
  _Float16* WkT = ws + off; off += (size_t)HIDDEN * (NKV * HD);   // 4M
  _Float16* WvT = ws + off; off += (size_t)HIDDEN * (NKV * HD);   // 4M
  _Float16* WoT = ws + off; off += (size_t)(NH * HD) * HIDDEN;    // 16M
  _Float16* Qh  = ws + off; off += (size_t)S_LEN * (NH * HD);     // 8M
  _Float16* Kh  = ws + off; off += (size_t)S_LEN * (NKV * HD);    // 2M
  _Float16* Vh  = ws + off; off += (size_t)S_LEN * (NKV * HD);    // 2M
  _Float16* Vt  = ws + off; off += (size_t)NKV * HD * S_LEN;      // 2M
  _Float16* Ah  = ws + off; off += (size_t)S_LEN * (NH * HD);     // 8M

  const int TB = 256;
  // 1) f32 -> f16 conversions (weights transposed to [N][K])
  int nx = S_LEN * HIDDEN;
  cvt_f32_f16<<<(nx + TB - 1) / TB, TB, 0, stream>>>(X, Xh, nx);
  int nq = HIDDEN * NH * HD;
  cvt_transpose_f32_f16<<<(nq + TB - 1) / TB, TB, 0, stream>>>(Wq, WqT, HIDDEN, NH * HD);
  int nk = HIDDEN * NKV * HD;
  cvt_transpose_f32_f16<<<(nk + TB - 1) / TB, TB, 0, stream>>>(Wk, WkT, HIDDEN, NKV * HD);
  cvt_transpose_f32_f16<<<(nk + TB - 1) / TB, TB, 0, stream>>>(Wv, WvT, HIDDEN, NKV * HD);
  cvt_transpose_f32_f16<<<(nq + TB - 1) / TB, TB, 0, stream>>>(Wo, WoT, NH * HD, HIDDEN);

  // 2) projections (WMMA GEMMs, f16 out)
  gemm_f16_wmma<0><<<dim3((NH * HD) / 128, S_LEN / 64), 256, 0, stream>>>(
      Xh, WqT, Qh, S_LEN, NH * HD, HIDDEN);
  gemm_f16_wmma<0><<<dim3((NKV * HD) / 128, S_LEN / 64), 256, 0, stream>>>(
      Xh, WkT, Kh, S_LEN, NKV * HD, HIDDEN);
  gemm_f16_wmma<0><<<dim3((NKV * HD) / 128, S_LEN / 64), 256, 0, stream>>>(
      Xh, WvT, Vh, S_LEN, NKV * HD, HIDDEN);

  // 3) RoPE on Q and K (in place)
  int nrq = S_LEN * NH * 64;
  rope_kernel<<<(nrq + TB - 1) / TB, TB, 0, stream>>>(Qh, NH);
  int nrk = S_LEN * NKV * 64;
  rope_kernel<<<(nrk + TB - 1) / TB, TB, 0, stream>>>(Kh, NKV);

  // 4) V -> V^T [kv][d][s] for contiguous-K PV fragments
  int nv = S_LEN * NKV * HD;
  transpose_v<<<(nv + TB - 1) / TB, TB, 0, stream>>>(Vh, Vt);

  // 5) causal flash attention (async-LDS staged K/V)
  flash_attn<<<dim3(NH, S_LEN / 64), 128, 0, stream>>>(Qh, Kh, Vt, Ah);

  // 6) output projection, fp32 result straight into d_out
  gemm_f16_wmma<1><<<dim3(HIDDEN / 128, S_LEN / 64), 256, 0, stream>>>(
      Ah, WoT, (void*)d_out, S_LEN, HIDDEN, NH * HD);

  (void)in_sizes; (void)n_in; (void)out_size; (void)ws_size;
}
